// PnaAgg_34737695490532
// MI455X (gfx1250) — compile-verified, matrix operand order checked by reading, not run on hardware
//
#include <hip/hip_runtime.h>
#include <hip/hip_bf16.h>

#define NN    50000
#define NE    800000
#define DD    128
#define KTOT  1536            // 12 * D
#define AVGDL 2.8332133440562162f   // ln(17)
#define EPSV  1e-5f

typedef __attribute__((ext_vector_type(16))) __bf16 v16bf;
typedef __attribute__((ext_vector_type(8)))  float  v8f;

// ---- order-preserving float <-> uint encoding (for atomicMin/Max on u32) ----
__device__ __forceinline__ unsigned enc_f32(float f) {
  unsigned u = __float_as_uint(f);
  return (u & 0x80000000u) ? ~u : (u | 0x80000000u);
}
__device__ __forceinline__ float dec_f32(unsigned u) {
  return __uint_as_float((u & 0x80000000u) ? (u ^ 0x80000000u) : ~u);
}

// ------------------------------ init workspace -------------------------------
__global__ void pna_init(float* __restrict__ s1, float* __restrict__ s2,
                         unsigned* __restrict__ mn, unsigned* __restrict__ mx,
                         unsigned* __restrict__ cnt) {
  int t = blockIdx.x * blockDim.x + threadIdx.x;
  if (t < NN * DD) {
    s1[t] = 0.0f;
    s2[t] = 0.0f;
    mn[t] = 0xFFFFFFFFu;   // +max in encoded order
    mx[t] = 0x00000000u;   // -max in encoded order
  }
  if (t < NN) cnt[t] = 0u;
}

// ------------------------- edge scatter (one wave/edge) ----------------------
// lane handles 4 consecutive features (float4 -> global_load_b128), then
// native f32 add atomics + u32 min/max atomics into L2-resident aggregates.
__global__ void pna_scatter(const float* __restrict__ x, const int* __restrict__ index,
                            float* __restrict__ s1, float* __restrict__ s2,
                            unsigned* __restrict__ mn, unsigned* __restrict__ mx,
                            unsigned* __restrict__ cnt) {
  int wave = threadIdx.x >> 5;
  int lane = threadIdx.x & 31;
  int e = blockIdx.x * (blockDim.x >> 5) + wave;
  if (e >= NE) return;
  int idx = index[e];
  const float4 v = *reinterpret_cast<const float4*>(x + (size_t)e * DD + lane * 4);
  size_t base = (size_t)idx * DD + lane * 4;

  unsafeAtomicAdd(&s1[base + 0], v.x);
  unsafeAtomicAdd(&s1[base + 1], v.y);
  unsafeAtomicAdd(&s1[base + 2], v.z);
  unsafeAtomicAdd(&s1[base + 3], v.w);
  unsafeAtomicAdd(&s2[base + 0], v.x * v.x);
  unsafeAtomicAdd(&s2[base + 1], v.y * v.y);
  unsafeAtomicAdd(&s2[base + 2], v.z * v.z);
  unsafeAtomicAdd(&s2[base + 3], v.w * v.w);
  atomicMin(&mn[base + 0], enc_f32(v.x));
  atomicMin(&mn[base + 1], enc_f32(v.y));
  atomicMin(&mn[base + 2], enc_f32(v.z));
  atomicMin(&mn[base + 3], enc_f32(v.w));
  atomicMax(&mx[base + 0], enc_f32(v.x));
  atomicMax(&mx[base + 1], enc_f32(v.y));
  atomicMax(&mx[base + 2], enc_f32(v.z));
  atomicMax(&mx[base + 3], enc_f32(v.w));
  if (lane == 0) atomicAdd(&cnt[idx], 1u);
}

// -------- finalize: stats + PNA degree scalers folded into bf16 A12 ----------
// A12 row layout: [mean|min|max|std | amp*(...) | att*(...)]  -> [NN, 1536]
__global__ void pna_finalize(const float* __restrict__ s1, const float* __restrict__ s2,
                             const unsigned* __restrict__ mn, const unsigned* __restrict__ mx,
                             const unsigned* __restrict__ cnt,
                             __hip_bfloat16* __restrict__ A12) {
  int t = blockIdx.x * blockDim.x + threadIdx.x;
  if (t >= NN * DD) return;
  int n = t >> 7;
  int d = t & (DD - 1);

  unsigned c = cnt[n];
  float cf   = (c == 0u) ? 1.0f : (float)c;
  float mean = s1[t] / cf;
  float var  = s2[t] / cf - mean * mean;
  float sd   = sqrtf(fmaxf(var, 0.0f) + EPSV);
  float lo   = (c == 0u) ? 0.0f : dec_f32(mn[t]);
  float hi   = (c == 0u) ? 0.0f : dec_f32(mx[t]);
  float logd = logf(cf + 1.0f);
  float amp  = logd / AVGDL;
  float att  = AVGDL / logd;

  __hip_bfloat16* row = A12 + (size_t)n * KTOT;
  row[0 * DD + d]  = __float2bfloat16(mean);
  row[1 * DD + d]  = __float2bfloat16(lo);
  row[2 * DD + d]  = __float2bfloat16(hi);
  row[3 * DD + d]  = __float2bfloat16(sd);
  row[4 * DD + d]  = __float2bfloat16(mean * amp);
  row[5 * DD + d]  = __float2bfloat16(lo * amp);
  row[6 * DD + d]  = __float2bfloat16(hi * amp);
  row[7 * DD + d]  = __float2bfloat16(sd * amp);
  row[8 * DD + d]  = __float2bfloat16(mean * att);
  row[9 * DD + d]  = __float2bfloat16(lo * att);
  row[10 * DD + d] = __float2bfloat16(hi * att);
  row[11 * DD + d] = __float2bfloat16(sd * att);
}

// -------------------------- W fp32 -> bf16 (393 KB) --------------------------
__global__ void pna_cvt_w(const float* __restrict__ W, __hip_bfloat16* __restrict__ Wb) {
  int t = blockIdx.x * blockDim.x + threadIdx.x;
  if (t < DD * KTOT) Wb[t] = __float2bfloat16(W[t]);
}

// ------------------------------- WMMA GEMM -----------------------------------
// out[M,128] = A12[M,1536](bf16) @ Wb[128,1536]^T(bf16) + bias, fp32 accumulate
// One block = one 16-row M tile; 8 waves cover the 8 N-tiles of 16.
// Per-lane bf16 fragment = two contiguous 16B chunks matching the ISA layout:
//   lane<16 : row = lane,    K = {k0..k0+7, k0+16..k0+23}
//   lane>=16: row = lane-16, K = {k0+8..k0+15, k0+24..k0+31}
__global__ void __launch_bounds__(256)
pna_gemm(const __hip_bfloat16* __restrict__ A, const __hip_bfloat16* __restrict__ Bw,
         const float* __restrict__ bias, float* __restrict__ out) {
  union Frag { uint4 q[2]; v16bf v; };

  int wave = threadIdx.x >> 5;
  int lane = threadIdx.x & 31;
  int m0 = blockIdx.x * 16;          // 50000 / 16 = 3125 exact
  int n0 = wave * 16;                // 8 waves * 16 = 128 cols
  int row  = lane & 15;
  int koff = (lane >> 4) * 8;

  const __hip_bfloat16* aptr = A  + (size_t)(m0 + row) * KTOT + koff;
  const __hip_bfloat16* bptr = Bw + (size_t)(n0 + row) * KTOT + koff;

  v8f acc = {};
  #pragma unroll 4
  for (int k0 = 0; k0 < KTOT; k0 += 32) {
    Frag a, b;
    a.q[0] = *reinterpret_cast<const uint4*>(aptr + k0);
    a.q[1] = *reinterpret_cast<const uint4*>(aptr + k0 + 16);
    b.q[0] = *reinterpret_cast<const uint4*>(bptr + k0);
    b.q[1] = *reinterpret_cast<const uint4*>(bptr + k0 + 16);
    acc = __builtin_amdgcn_wmma_f32_16x16x32_bf16(
        /*neg_a=*/false, a.v, /*neg_b=*/false, b.v,
        /*c_mod=*/(short)0, acc, /*reuse_a=*/false, /*reuse_b=*/false);
  }

  int col  = n0 + (lane & 15);
  int mrow = m0 + ((lane >> 4) << 3);   // C layout: VGPR r -> M = r (+8 for hi lanes)
  float bv = bias[col];
  #pragma unroll
  for (int r = 0; r < 8; ++r) {
    out[(size_t)(mrow + r) * DD + col] = acc[r] + bv;
  }
}

// ------------------------------- launcher ------------------------------------
extern "C" void kernel_launch(void* const* d_in, const int* in_sizes, int n_in,
                              void* d_out, int out_size, void* d_ws, size_t ws_size,
                              hipStream_t stream) {
  const float* x     = (const float*)d_in[0];   // [800000,128]
  const int*   index = (const int*)d_in[1];     // [800000]
  const float* W     = (const float*)d_in[2];   // [128,1536]
  const float* bias  = (const float*)d_in[3];   // [128]
  float*       out   = (float*)d_out;           // [50000,128]

  // workspace carve-up (256B aligned)
  char* ws = (char*)d_ws;
  size_t off = 0;
  auto take = [&](size_t bytes) {
    char* p = ws + off;
    off += (bytes + 255) & ~(size_t)255;
    return p;
  };
  float*          s1  = (float*)take((size_t)NN * DD * 4);     // 25.6 MB
  float*          s2  = (float*)take((size_t)NN * DD * 4);     // 25.6 MB
  unsigned*       mn  = (unsigned*)take((size_t)NN * DD * 4);  // 25.6 MB
  unsigned*       mx  = (unsigned*)take((size_t)NN * DD * 4);  // 25.6 MB
  unsigned*       cnt = (unsigned*)take((size_t)NN * 4);       // 0.2 MB
  __hip_bfloat16* Wb  = (__hip_bfloat16*)take((size_t)DD * KTOT * 2);   // 0.4 MB
  __hip_bfloat16* A12 = (__hip_bfloat16*)take((size_t)NN * KTOT * 2);   // 153.6 MB
  (void)ws_size; (void)n_in; (void)in_sizes; (void)out_size;

  // 1) init aggregates
  {
    int nt = NN * DD;
    pna_init<<<(nt + 255) / 256, 256, 0, stream>>>(s1, s2, mn, mx, cnt);
  }
  // 2) atomic scatter: one wave32 per edge, 8 edges per 256-thread block
  {
    int blocks = (NE + 7) / 8;
    pna_scatter<<<blocks, 256, 0, stream>>>(x, index, s1, s2, mn, mx, cnt);
  }
  // 3) finalize stats -> bf16 A12 with PNA scalers folded in
  {
    int nt = NN * DD;
    pna_finalize<<<(nt + 255) / 256, 256, 0, stream>>>(s1, s2, mn, mx, cnt, A12);
  }
  // 4) W -> bf16
  {
    int nt = DD * KTOT;
    pna_cvt_w<<<(nt + 255) / 256, 256, 0, stream>>>(W, Wb);
  }
  // 5) WMMA GEMM + bias
  {
    pna_gemm<<<NN / 16, 256, 0, stream>>>(A12, Wb, bias, out);
  }
}
